// GNNSat_V3_18940805776103
// MI455X (gfx1250) — matrix operand order repaired and checked by reading, not compiled, not run on hardware
//
#include <hip/hip_runtime.h>
#include <math.h>

// ---------------------------------------------------------------------------
// GNN-SAT GAT pipeline for gfx1250 (MI455X).  N=50000, E=800000, H=64, 3 main
// layers.  Dense transforms use v_wmma_f32_16x16x32_f16 with pre-packed f16
// B operands (b128 loads only); edge phase uses L2-resident float atomics
// (segment softmax).  wave32 throughout.
// ---------------------------------------------------------------------------

#define BLK 256
#define HD  64
#define NSLA 0.2f   // attention leaky slope
#define NSLACT 0.01f
#define BNEPS 1e-5f
#define WPACK_PER_LAYER 4096   // 2 kk * 4 nt * 32 lanes * 16 halfs

typedef __attribute__((ext_vector_type(16))) _Float16 v16h;
typedef __attribute__((ext_vector_type(8)))  float    v8f;

__device__ __forceinline__ int f2ord(float f) {
  int i = __float_as_int(f);
  return i >= 0 ? i : (i ^ 0x7FFFFFFF);           // monotone float->int key
}
__device__ __forceinline__ float ord2f(int i) {
  return __int_as_float(i >= 0 ? i : (i ^ 0x7FFFFFFF));
}
__device__ __forceinline__ float lrelu(float x, float s) {
  return x >= 0.f ? x : s * x;
}

// ------------------------------- utility fills ------------------------------
__global__ void k_fill_f32(float* p, float v, int n) {
  int i = blockIdx.x * blockDim.x + threadIdx.x;
  if (i < n) p[i] = v;
}
__global__ void k_fill_i32(int* p, int v, int n) {
  int i = blockIdx.x * blockDim.x + threadIdx.x;
  if (i < n) p[i] = v;
}
// hout[n][c] = bias[c]  (aggregation then atomically adds on top)
__global__ void k_bias_init(float* h, const float* __restrict__ b, int total) {
  int i = blockIdx.x * blockDim.x + threadIdx.x;
  if (i < total) h[i] = b[i & (HD - 1)];
}

// --------------------- degree + mean incoming edge_attr ---------------------
__global__ void k_deg(const int* __restrict__ ei, const float* __restrict__ ea,
                      float* deg, float* easum, int E_) {
  int e = blockIdx.x * blockDim.x + threadIdx.x;
  if (e >= E_) return;
  int d = ei[E_ + e];                               // row 1 = dst
  atomicAdd(&deg[d], 1.0f);
  atomicAdd(&easum[d * 2 + 0], ea[e * 2 + 0]);
  atomicAdd(&easum[d * 2 + 1], ea[e * 2 + 1]);
}
__global__ void k_ea_mean(float* eams, const float* __restrict__ deg, int N_) {
  int n = blockIdx.x * blockDim.x + threadIdx.x;
  if (n >= N_) return;
  float dg = deg[n];
  dg = dg < 1.0f ? 1.0f : dg;
  eams[n * 2 + 0] /= dg;
  eams[n * 2 + 1] /= dg;
}

// ------------- layer-0 transform (K=2) fused with score dots ---------------
__global__ void k_l0_gemm(const float* __restrict__ x, const float* __restrict__ W0,
                          const float* __restrict__ a_s, const float* __restrict__ a_d,
                          float* __restrict__ ht, float* __restrict__ ssrc,
                          float* __restrict__ sdst, int N_) {
  int n = blockIdx.x * blockDim.x + threadIdx.x;
  if (n >= N_) return;
  float x0 = x[n * 2 + 0], x1 = x[n * 2 + 1];
  float s1 = 0.f, s2 = 0.f;
  #pragma unroll 8
  for (int j = 0; j < HD; ++j) {
    float v = fmaf(x0, W0[j], x1 * W0[HD + j]);
    ht[n * HD + j] = v;
    s1 = fmaf(v, a_s[j], s1);
    s2 = fmaf(v, a_d[j], s2);
  }
  ssrc[n] = s1;
  sdst[n] = s2;
}

// ---- pack all main-layer weights into WMMA B-operand f16 lane layout ------
// Wp[layer][kk][nt][lane][j] = (f16) W[layer][ (kk*32 + (lane>=16)*16 + j) ][ nt*16 + (lane&15) ]
__global__ void k_pack_w(const float* __restrict__ W, _Float16* __restrict__ Wp,
                         int total) {
  int i = blockIdx.x * blockDim.x + threadIdx.x;
  if (i >= total) return;
  int layer = i / WPACK_PER_LAYER;
  int r = i - layer * WPACK_PER_LAYER;
  int j    = r & 15;
  int lane = (r >> 4) & 31;
  int nt   = (r >> 9) & 3;
  int kk   = r >> 11;
  int k    = kk * 32 + ((lane >> 4) << 4) + j;
  int col  = nt * 16 + (lane & 15);
  Wp[i] = (_Float16)W[(long long)layer * HD * HD + k * HD + col];
}

// -------------------- WMMA GEMM:  ht[N,64] = hin[N,64] @ W[64,64] ----------
// One wave per 16-row tile; 2 K-blocks of 32 x 4 N-tiles of 16 -> 8 WMMAs.
// A: 4x b128 f32 loads + pack; B: pre-packed f16, 2x b128 loads per fragment.
__global__ void k_gemm_wmma(const float* __restrict__ hin,
                            const _Float16* __restrict__ Wp,
                            float* __restrict__ ht, int nrows) {
  int wave = (blockIdx.x * blockDim.x + threadIdx.x) >> 5;
  int lane = threadIdx.x & 31;
  int m0 = wave * 16;
  if (m0 >= nrows) return;                          // wave-uniform: EXEC stays all-1s
  int hi = lane >> 4;                               // 0 / 1 lane half
  int l15 = lane & 15;
  v8f zero = {};
  v8f acc0 = zero, acc1 = zero, acc2 = zero, acc3 = zero;

  const float* arow = hin + (m0 + l15) * HD;
  #pragma unroll
  for (int kk = 0; kk < 2; ++kk) {
    // A fragment: lanes 0-15 need K = kbA..kbA+15 (contiguous, four
    // float4s), lanes 16-31 the same shifted by 16.
    int kbA = kk * 32 + hi * 16;
    const float4* ap0 = (const float4*)(arow + kbA);        // 32B aligned
    const float4* ap1 = (const float4*)(arow + kbA + 8);
    float4 a0 = ap0[0], a1 = ap0[1], a2 = ap1[0], a3 = ap1[1];
    v16h a;
    a[0]  = (_Float16)a0.x; a[1]  = (_Float16)a0.y;
    a[2]  = (_Float16)a0.z; a[3]  = (_Float16)a0.w;
    a[4]  = (_Float16)a1.x; a[5]  = (_Float16)a1.y;
    a[6]  = (_Float16)a1.z; a[7]  = (_Float16)a1.w;
    a[8]  = (_Float16)a2.x; a[9]  = (_Float16)a2.y;
    a[10] = (_Float16)a2.z; a[11] = (_Float16)a2.w;
    a[12] = (_Float16)a3.x; a[13] = (_Float16)a3.y;
    a[14] = (_Float16)a3.z; a[15] = (_Float16)a3.w;

    #pragma unroll
    for (int nt = 0; nt < 4; ++nt) {
      // packed B fragment: one contiguous 32B region per lane
      v16h b = *(const v16h*)(Wp + (((kk * 4 + nt) * 32) + lane) * 16);
      v8f c = (nt == 0) ? acc0 : (nt == 1) ? acc1 : (nt == 2) ? acc2 : acc3;
      c = __builtin_amdgcn_wmma_f32_16x16x32_f16(false, a, false, b, (short)0, c,
                                                 false, false);
      if (nt == 0) acc0 = c; else if (nt == 1) acc1 = c;
      else if (nt == 2) acc2 = c; else acc3 = c;
    }
  }
  // C/D layout: VGPR r -> M = r + 8*hi, N = l15 (+16 per tile)
  #pragma unroll
  for (int nt = 0; nt < 4; ++nt) {
    v8f c = (nt == 0) ? acc0 : (nt == 1) ? acc1 : (nt == 2) ? acc2 : acc3;
    int col = nt * 16 + l15;
    #pragma unroll
    for (int r = 0; r < 8; ++r) {
      int row = m0 + r + hi * 8;
      ht[row * HD + col] = c[r];
    }
  }
}

// ------------------- per-node attention score dots -------------------------
__global__ void k_scores(const float* __restrict__ ht, const float* __restrict__ a_s,
                         const float* __restrict__ a_d, float* ssrc, float* sdst,
                         int N_) {
  int n = blockIdx.x * blockDim.x + threadIdx.x;
  if (n >= N_) return;
  const float* r = ht + n * HD;
  float s1 = 0.f, s2 = 0.f;
  #pragma unroll 8
  for (int j = 0; j < HD; ++j) {
    float v = r[j];
    s1 = fmaf(v, a_s[j], s1);
    s2 = fmaf(v, a_d[j], s2);
  }
  ssrc[n] = s1;
  sdst[n] = s2;
}

// we2[d] = sum_h We[d][h]*ae[h]   (2 tiny dot products)
__global__ void k_we2(const float* __restrict__ We, const float* __restrict__ ae,
                      float* we2) {
  float s0 = 0.f, s1 = 0.f;
  for (int j = 0; j < HD; ++j) {
    s0 = fmaf(We[j], ae[j], s0);
    s1 = fmaf(We[HD + j], ae[j], s1);
  }
  we2[0] = s0;
  we2[1] = s1;
}

// ---------------------------- edge passes ----------------------------------
__global__ void k_alpha_max(const int* __restrict__ ei, const float* __restrict__ ea,
                            const float* __restrict__ eamean,
                            const float* __restrict__ ssrc, const float* __restrict__ sdst,
                            const float* __restrict__ we2, float* __restrict__ alpha,
                            int* __restrict__ mEnc, int E_, int N_) {
  int e = blockIdx.x * blockDim.x + threadIdx.x;
  if (e >= E_ + N_) return;
  int s, d; float e0, e1;
  if (e < E_) { s = ei[e]; d = ei[E_ + e]; e0 = ea[e * 2]; e1 = ea[e * 2 + 1]; }
  else { s = d = e - E_; e0 = eamean[s * 2]; e1 = eamean[s * 2 + 1]; }
  float a = ssrc[s] + sdst[d] + e0 * we2[0] + e1 * we2[1];
  a = lrelu(a, NSLA);
  alpha[e] = a;
  atomicMax(&mEnc[d], f2ord(a));
}

__global__ void k_expsum(const int* __restrict__ ei, float* __restrict__ alpha,
                         const int* __restrict__ mEnc, float* __restrict__ ssum,
                         int E_, int N_) {
  int e = blockIdx.x * blockDim.x + threadIdx.x;
  if (e >= E_ + N_) return;
  int d = (e < E_) ? ei[E_ + e] : (e - E_);
  float ex = __expf(alpha[e] - ord2f(mEnc[d]));
  alpha[e] = ex;                                    // reuse buffer for exp()
  atomicAdd(&ssum[d], ex);
}

// 16 threads per edge, 4 channels each (float4 gather, 4 scatter atomics)
__global__ void k_aggregate(const int* __restrict__ ei, const float* __restrict__ exb,
                            const float* __restrict__ ssum, const float* __restrict__ ht,
                            float* __restrict__ hout, int E_, int N_) {
  long long tid = (long long)blockIdx.x * blockDim.x + threadIdx.x;
  int e = (int)(tid >> 4);
  if (e >= E_ + N_) return;
  int c = ((int)tid & 15) * 4;
  int s, d;
  if (e < E_) { s = ei[e]; d = ei[E_ + e]; } else { s = d = e - E_; }
  float coef = exb[e] / (ssum[d] + 1e-16f);
  const float4 hv = *(const float4*)(ht + s * HD + c);
  float* o = hout + d * HD + c;
  atomicAdd(o + 0, coef * hv.x);
  atomicAdd(o + 1, coef * hv.y);
  atomicAdd(o + 2, coef * hv.z);
  atomicAdd(o + 3, coef * hv.w);
}

// ---------------------------- batchnorm ------------------------------------
__global__ void k_bn_stats(const float* __restrict__ h, float* sums, int N_) {
  __shared__ float ls[HD], lss[HD];
  int t = threadIdx.x;
  if (t < HD) { ls[t] = 0.f; lss[t] = 0.f; }
  __syncthreads();
  int c = t & (HD - 1);
  int start = blockIdx.x * 256;
  int end = start + 256; if (end > N_) end = N_;
  float s = 0.f, ss = 0.f;
  for (int n = start + (t >> 6); n < end; n += 4) {
    float v = h[n * HD + c];
    s += v; ss += v * v;
  }
  atomicAdd(&ls[c], s);
  atomicAdd(&lss[c], ss);
  __syncthreads();
  if (t < HD) { atomicAdd(&sums[t], ls[t]); atomicAdd(&sums[HD + t], lss[t]); }
}

__global__ void k_bn_apply(float* __restrict__ h, const float* __restrict__ sums,
                           const float* __restrict__ gamma, const float* __restrict__ beta,
                           int N_) {
  int i = blockIdx.x * blockDim.x + threadIdx.x;
  if (i >= N_ * HD) return;
  int c = i & (HD - 1);
  float inv_n = 1.0f / (float)N_;
  float mu = sums[c] * inv_n;
  float var = sums[HD + c] * inv_n - mu * mu;
  float v = gamma[c] * (h[i] - mu) * rsqrtf(var + BNEPS) + beta[c];
  h[i] = lrelu(v, NSLACT);
}

__global__ void k_act(float* h, int total) {
  int i = blockIdx.x * blockDim.x + threadIdx.x;
  if (i < total) h[i] = lrelu(h[i], NSLACT);
}

// ------------------------------ final MLP ----------------------------------
__global__ void k_final(const float* __restrict__ h, const float* __restrict__ Wf1,
                        const float* __restrict__ bf1, const float* __restrict__ Wf2,
                        const float* __restrict__ bf2, const float* __restrict__ mask,
                        float* __restrict__ out, int N_) {
  int n = blockIdx.x * blockDim.x + threadIdx.x;
  if (n >= N_) return;
  float hr[HD];
  #pragma unroll
  for (int k = 0; k < HD; ++k) hr[k] = h[n * HD + k];
  float o = bf2[0];
  for (int j = 0; j < 32; ++j) {
    float acc = bf1[j];
    #pragma unroll
    for (int k = 0; k < HD; ++k) acc = fmaf(hr[k], Wf1[k * 32 + j], acc);
    o = fmaf(lrelu(acc, NSLACT), Wf2[j], o);
  }
  out[n] = o * mask[n];
}

// ---------------------------------------------------------------------------
static inline int cdiv(long long a, long long b) { return (int)((a + b - 1) / b); }

static void edge_phase(const int* ei, const float* eattr, const float* eamean,
                       const float* ht, const float* a_s, const float* a_d,
                       const float* We, const float* ae, const float* bias,
                       float* ssrc, float* sdst, float* we2, int* mEnc,
                       float* ssum, float* exbuf, float* hout,
                       int E_, int N_, bool layer0, hipStream_t st) {
  if (!layer0)
    k_scores<<<cdiv(N_, BLK), BLK, 0, st>>>(ht, a_s, a_d, ssrc, sdst, N_);
  k_we2<<<1, 1, 0, st>>>(We, ae, we2);
  k_fill_i32<<<cdiv(N_, BLK), BLK, 0, st>>>(mEnc, (int)0x807FFFFF, N_); // ord(-inf)
  k_fill_f32<<<cdiv(N_, BLK), BLK, 0, st>>>(ssum, 0.f, N_);
  k_bias_init<<<cdiv((long long)N_ * HD, BLK), BLK, 0, st>>>(hout, bias, N_ * HD);
  int Et = E_ + N_;
  k_alpha_max<<<cdiv(Et, BLK), BLK, 0, st>>>(ei, eattr, eamean, ssrc, sdst, we2,
                                             exbuf, mEnc, E_, N_);
  k_expsum<<<cdiv(Et, BLK), BLK, 0, st>>>(ei, exbuf, mEnc, ssum, E_, N_);
  k_aggregate<<<cdiv((long long)Et * 16, BLK), BLK, 0, st>>>(ei, exbuf, ssum, ht,
                                                             hout, E_, N_);
}

extern "C" void kernel_launch(void* const* d_in, const int* in_sizes, int n_in,
                              void* d_out, int out_size, void* d_ws, size_t ws_size,
                              hipStream_t stream) {
  const float* x      = (const float*)d_in[0];
  const int*   ei     = (const int*)d_in[1];
  const float* eattr  = (const float*)d_in[2];
  const float* mask   = (const float*)d_in[3];
  const float* W0     = (const float*)d_in[4];
  const float* as0    = (const float*)d_in[5];
  const float* ad0    = (const float*)d_in[6];
  const float* We0    = (const float*)d_in[7];
  const float* ae0    = (const float*)d_in[8];
  const float* b0     = (const float*)d_in[9];
  const float* Wm     = (const float*)d_in[10];
  const float* asm_   = (const float*)d_in[11];
  const float* adm    = (const float*)d_in[12];
  const float* Wem    = (const float*)d_in[13];
  const float* aem    = (const float*)d_in[14];
  const float* bm     = (const float*)d_in[15];
  const float* gamma  = (const float*)d_in[16];
  const float* beta   = (const float*)d_in[17];
  const float* Wf1    = (const float*)d_in[18];
  const float* bf1    = (const float*)d_in[19];
  const float* Wf2    = (const float*)d_in[20];
  const float* bf2    = (const float*)d_in[21];

  const int N = in_sizes[3];            // mask is [N]
  const int E = in_sizes[2] / 2;        // edge_attr is [E,2]
  const int Et = E + N;
  const int ITER = 3;

  // workspace carve-up (all offsets multiple of 8 floats -> 32B aligned)
  float* ws    = (float*)d_ws;
  float* deg   = ws;                ws += N;
  float* eams  = ws;                ws += 2 * N;   // easum -> in-place mean
  float* hA    = ws;                ws += (long long)N * HD;
  float* hB    = ws;                ws += (long long)N * HD;
  float* hT    = ws;                ws += (long long)N * HD;
  float* ssrc  = ws;                ws += N;
  float* sdst  = ws;                ws += N;
  int*   mEnc  = (int*)ws;          ws += N;
  float* ssum  = ws;                ws += N;
  float* exbuf = ws;                ws += Et;
  float* we2   = ws;                ws += 8;
  float* bnsum = ws;                ws += 2 * HD;
  _Float16* wpack = (_Float16*)ws;  ws += (ITER * WPACK_PER_LAYER) / 2;

  // ---- pack main-layer GEMM weights into WMMA B lane layout (f16) ----
  k_pack_w<<<cdiv(ITER * WPACK_PER_LAYER, BLK), BLK, 0, stream>>>(
      Wm, wpack, ITER * WPACK_PER_LAYER);

  // ---- self-loop edge_attr: mean of incoming edge_attr per node ----
  k_fill_f32<<<cdiv(3LL * N, BLK), BLK, 0, stream>>>(deg, 0.f, 3 * N); // deg + easum
  k_deg<<<cdiv(E, BLK), BLK, 0, stream>>>(ei, eattr, deg, eams, E);
  k_ea_mean<<<cdiv(N, BLK), BLK, 0, stream>>>(eams, deg, N);

  // ---- layer 0 (in=2): transform fused with score dots ----
  k_l0_gemm<<<cdiv(N, BLK), BLK, 0, stream>>>(x, W0, as0, ad0, hT, ssrc, sdst, N);
  edge_phase(ei, eattr, eams, hT, as0, ad0, We0, ae0, b0,
             ssrc, sdst, we2, mEnc, ssum, exbuf, hB, E, N, /*layer0=*/true, stream);

  // ---- batchnorm (training-mode batch stats) + leaky ----
  k_fill_f32<<<1, 2 * HD, 0, stream>>>(bnsum, 0.f, 2 * HD);
  k_bn_stats<<<cdiv(N, 256), 256, 0, stream>>>(hB, bnsum, N);
  k_bn_apply<<<cdiv((long long)N * HD, BLK), BLK, 0, stream>>>(hB, bnsum, gamma, beta, N);

  // ---- 3 main GAT layers: WMMA transform + edge phase, ping-pong hA/hB ----
  float* hin = hB;
  float* hout = hA;
  const int gemmGrid = cdiv((long long)cdiv(N, 16) * 32, BLK); // one wave / 16 rows
  for (int i = 0; i < ITER; ++i) {
    k_gemm_wmma<<<gemmGrid, BLK, 0, stream>>>(hin, wpack + (long long)i * WPACK_PER_LAYER,
                                              hT, N);
    edge_phase(ei, eattr, eams, hT, asm_ + i * HD, adm + i * HD,
               Wem + i * 2 * HD, aem + i * HD, bm + i * HD,
               ssrc, sdst, we2, mEnc, ssum, exbuf, hout, E, N, false, stream);
    k_act<<<cdiv((long long)N * HD, BLK), BLK, 0, stream>>>(hout, N * HD);
    float* t = hin; hin = hout; hout = t;
  }
  // after 3 swaps, final features live in `hin`

  // ---- final MLP + mask ----
  k_final<<<cdiv(N, BLK), BLK, 0, stream>>>(hin, Wf1, bf1, Wf2, bf2, mask,
                                            (float*)d_out, N);
}
